// TransformerBlock_45990509806234
// MI455X (gfx1250) — matrix operand order repaired and checked by reading, not compiled, not run on hardware
//
#include <hip/hip_runtime.h>
#include <hip/hip_bf16.h>
#include <math.h>

// ---------------------------------------------------------------------------
// Problem constants (from reference)
// ---------------------------------------------------------------------------
#define DMODEL 1024
#define NHEADS 16
#define HEADD  64
#define DFFN   4096
#define BATCH  2
#define SEQL   2048
#define MTOT   (BATCH * SEQL)   // 4096 rows

// ---------------------------------------------------------------------------
// WMMA types / helpers (CDNA5 gfx1250, wave32)
// ---------------------------------------------------------------------------
typedef __attribute__((ext_vector_type(16))) __bf16          v16bf;
typedef __attribute__((ext_vector_type(16))) unsigned short  v16us;
typedef __attribute__((ext_vector_type(8)))  float           v8f;
typedef __attribute__((ext_vector_type(4)))  unsigned int    v4u;

struct u32x8 { v4u lo, hi; };

// Load a 16-element bf16 fragment as two aligned 16B chunks (-> ds/global b128)
static __device__ __forceinline__ v16us ld_pair128(const unsigned short* p0,
                                                   const unsigned short* p1) {
  u32x8 t;
  t.lo = *reinterpret_cast<const v4u*>(p0);
  t.hi = *reinterpret_cast<const v4u*>(p1);
  return __builtin_bit_cast(v16us, t);
}

// D = A(16x32 bf16) * B(32x16 bf16) + C(16x16 f32)
static __device__ __forceinline__ v8f wmma_bf16(v16us a, v16us b, v8f c) {
  return __builtin_amdgcn_wmma_f32_16x16x32_bf16(
      false, __builtin_bit_cast(v16bf, a),
      false, __builtin_bit_cast(v16bf, b),
      (short)0, c, false, false);
}

// float -> bf16 bits, round-to-nearest-even
static __device__ __forceinline__ unsigned short f2bf(float f) {
  unsigned int u = __float_as_uint(f);
  u += 0x7FFFu + ((u >> 16) & 1u);
  return (unsigned short)(u >> 16);
}

// ---------------------------------------------------------------------------
// f32 -> bf16 conversion (grid-stride)
// ---------------------------------------------------------------------------
__global__ void cvt_bf16_kernel(const float* __restrict__ in,
                                unsigned short* __restrict__ out, int n) {
  int i = blockIdx.x * blockDim.x + threadIdx.x;
  int stride = gridDim.x * blockDim.x;
  for (; i < n; i += stride) out[i] = f2bf(in[i]);
}

// ---------------------------------------------------------------------------
// affine = t_emb @ aff_w + aff_b        out: [BATCH, 2*DMODEL] f32
// ---------------------------------------------------------------------------
__global__ __launch_bounds__(256) void affine_kernel(
    const float* __restrict__ t_emb, const float* __restrict__ w,
    const float* __restrict__ bias, float* __restrict__ out) {
  int idx = blockIdx.x * blockDim.x + threadIdx.x;   // over BATCH*2*DMODEL
  int b = idx / (2 * DMODEL);
  int j = idx - b * 2 * DMODEL;
  float acc = bias[j];
  const float* te = t_emb + (size_t)b * DMODEL;
  for (int k = 0; k < DMODEL; k++) acc += te[k] * w[(size_t)k * (2 * DMODEL) + j];
  out[idx] = acc;
}

// ---------------------------------------------------------------------------
// AdaLN: out_bf16 = gamma * ((x-mu)*rsqrt(var+eps)*g + b) + beta, per row
// ---------------------------------------------------------------------------
__global__ __launch_bounds__(256) void adaln_kernel(
    const float* __restrict__ x, const float* __restrict__ g,
    const float* __restrict__ bb, const float* __restrict__ aff,
    unsigned short* __restrict__ out) {
  __shared__ float shs[8], shss[8];
  const int row = blockIdx.x;
  const int b = row / SEQL;
  const int tid = threadIdx.x;
  const float* xr = x + (size_t)row * DMODEL;
  float vals[4];
  float s = 0.0f, ss = 0.0f;
#pragma unroll
  for (int i = 0; i < 4; i++) {
    float v = xr[tid + i * 256];
    vals[i] = v; s += v; ss += v * v;
  }
#pragma unroll
  for (int off2 = 1; off2 < 32; off2 <<= 1) {
    s += __shfl_xor(s, off2, 32);
    ss += __shfl_xor(ss, off2, 32);
  }
  if ((tid & 31) == 0) { shs[tid >> 5] = s; shss[tid >> 5] = ss; }
  __syncthreads();
  float ts = 0.0f, tss = 0.0f;
#pragma unroll
  for (int i = 0; i < 8; i++) { ts += shs[i]; tss += shss[i]; }
  float mu  = ts * (1.0f / DMODEL);
  float var = tss * (1.0f / DMODEL) - mu * mu;
  float inv = rsqrtf(var + 1e-5f);
  const float* affb = aff + (size_t)b * (2 * DMODEL);
#pragma unroll
  for (int i = 0; i < 4; i++) {
    int c = tid + i * 256;
    float normed = (vals[i] - mu) * inv * g[c] + bb[c];
    float vv = affb[c] * normed + affb[DMODEL + c];
    out[(size_t)row * DMODEL + c] = f2bf(vv);
  }
}

// ---------------------------------------------------------------------------
// WMMA GEMM: C[M,N] = A_bf16[M,K] @ W_bf16[K,N]  (+bias, gelu, +residual)
// block = 256 thr (8 waves), block tile 128x128, wave tile 32x64, K step 32.
// A staged row-major (fragment rows contiguous); B staged TRANSPOSED in LDS
// so B fragments are two aligned b128 loads per lane. Register-prefetch
// pipelining: tile k+1 is in flight while tile k computes.
// ---------------------------------------------------------------------------
__global__ __launch_bounds__(256) void gemm_bf16_kernel(
    const unsigned short* __restrict__ A, const unsigned short* __restrict__ W,
    const float* __restrict__ bias, const float* residual,
    float* outF, unsigned short* outH,
    int M, int N, int K, int doGelu) {
  __shared__ unsigned short Asub[128][40];  // [m][k], 16B-aligned rows
  __shared__ unsigned short Bt[128][40];    // [n][k] (transposed tile)
  const int tid  = threadIdx.x;
  const int lane = tid & 31, wid = tid >> 5;
  const int lm = lane & 15, lh = lane >> 4;
  const int wm = wid & 3, wn = wid >> 2;
  const int mBlock = blockIdx.y * 128;
  const int nBlock = blockIdx.x * 128;
  (void)M;

  v8f zero = {};
  v8f acc[2][4];
#pragma unroll
  for (int mi = 0; mi < 2; mi++)
#pragma unroll
    for (int ni = 0; ni < 4; ni++) acc[mi][ni] = zero;

  unsigned int aReg[8], b0Reg[4], b1Reg[4];

  auto loadTile = [&](int kb) {
#pragma unroll
    for (int t = 0; t < 8; t++) {           // A: 128x32 = 2048 dwords
      int i = tid + t * 256;
      int r = i >> 4, cp = i & 15;
      aReg[t] = *reinterpret_cast<const unsigned int*>(
          &A[(size_t)(mBlock + r) * K + kb + cp * 2]);
    }
#pragma unroll
    for (int t = 0; t < 4; t++) {           // B: 32x128 as 2x2 blocks
      int i = tid + t * 256;
      int br = (i >> 6) * 2, bc = (i & 63) * 2;
      b0Reg[t] = *reinterpret_cast<const unsigned int*>(
          &W[(size_t)(kb + br) * N + nBlock + bc]);
      b1Reg[t] = *reinterpret_cast<const unsigned int*>(
          &W[(size_t)(kb + br + 1) * N + nBlock + bc]);
    }
  };
  auto storeTile = [&]() {
#pragma unroll
    for (int t = 0; t < 8; t++) {
      int i = tid + t * 256;
      int r = i >> 4, cp = i & 15;
      *reinterpret_cast<unsigned int*>(&Asub[r][cp * 2]) = aReg[t];
    }
#pragma unroll
    for (int t = 0; t < 4; t++) {           // 2x2 register transpose
      int i = tid + t * 256;
      int br = (i >> 6) * 2, bc = (i & 63) * 2;
      unsigned int d0 = b0Reg[t], d1 = b1Reg[t];
      unsigned int c0 = (d0 & 0xFFFFu) | (d1 << 16);
      unsigned int c1 = (d0 >> 16) | (d1 & 0xFFFF0000u);
      *reinterpret_cast<unsigned int*>(&Bt[bc][br])     = c0;
      *reinterpret_cast<unsigned int*>(&Bt[bc + 1][br]) = c1;
    }
  };

  loadTile(0);
  for (int kb = 0; kb < K; kb += 32) {
    __syncthreads();
    storeTile();
    __syncthreads();
    if (kb + 32 < K) loadTile(kb + 32);
    if (kb + 64 < K) {                      // L2 warm: global_prefetch_b8
      __builtin_prefetch(&A[(size_t)(mBlock + (tid & 127)) * K + kb + 64], 0, 1);
      if (tid < 32)
        __builtin_prefetch(&W[(size_t)(kb + 64 + tid) * N + nBlock], 0, 1);
    }

    v16us af[2], bfr[4];
#pragma unroll
    for (int mi = 0; mi < 2; mi++) {
      int mr = wm * 32 + mi * 16 + lm;
      af[mi] = ld_pair128(&Asub[mr][lh * 8], &Asub[mr][16 + lh * 8]);
    }
#pragma unroll
    for (int ni = 0; ni < 4; ni++) {
      int nc = wn * 64 + ni * 16 + lm;
      bfr[ni] = ld_pair128(&Bt[nc][lh * 8], &Bt[nc][16 + lh * 8]);
    }
#pragma unroll
    for (int mi = 0; mi < 2; mi++)
#pragma unroll
      for (int ni = 0; ni < 4; ni++)
        acc[mi][ni] = wmma_bf16(af[mi], bfr[ni], acc[mi][ni]);
  }

  // epilogue (C layout: VGPR r -> row lh*8+r, col = lm)
#pragma unroll
  for (int mi = 0; mi < 2; mi++) {
#pragma unroll
    for (int ni = 0; ni < 4; ni++) {
      int baseR = mBlock + wm * 32 + mi * 16 + lh * 8;
      int c = nBlock + wn * 64 + ni * 16 + lm;
      float bia = bias ? bias[c] : 0.0f;
#pragma unroll
      for (int r = 0; r < 8; r++) {
        float vv = acc[mi][ni][r] + bia;
        if (doGelu) vv = 0.5f * vv * (1.0f + erff(vv * 0.70710678118654752f));
        size_t idx = (size_t)(baseR + r) * N + c;
        if (residual) vv += residual[idx];
        if (outF) outF[idx] = vv;
        if (outH) outH[idx] = f2bf(vv);
      }
    }
  }
}

// ---------------------------------------------------------------------------
// Flash attention: Q,K,V bf16 [MTOT, DMODEL] (head h at cols h*64..h*64+63)
// grid (SEQL/128, BATCH*NHEADS), block 256 (8 waves); each wave owns 16 q-rows.
// K/V tiles (32 k-rows) are cooperatively staged in LDS once per chunk and
// shared by all 8 waves; all fragments are aligned b128 LDS/global loads.
// ---------------------------------------------------------------------------
__global__ __launch_bounds__(256) void flash_attn_kernel(
    const unsigned short* __restrict__ Q, const unsigned short* __restrict__ Km,
    const unsigned short* __restrict__ Vm, unsigned short* __restrict__ O) {
  __shared__ unsigned short Kt[32][72];        // [k-row][d] row-major
  __shared__ unsigned short Vt[64][40];        // [d][k-row] transposed
  __shared__ unsigned short Plds[8][16][40];   // per-wave P transpose buffer
  const int tid  = threadIdx.x;
  const int lane = tid & 31, wid = tid >> 5;
  const int lm = lane & 15, lh = lane >> 4;
  const int bh = blockIdx.y;
  const int b = bh / NHEADS, h = bh % NHEADS;
  const int qBase = blockIdx.x * 128 + wid * 16;
  const size_t rowQ0 = (size_t)b * SEQL + qBase;
  const int hcol = h * HEADD;

  // Q fragments (A layout): lane lm = row; two contiguous 8-elem runs per lane
  v16us qf[2];
#pragma unroll
  for (int c = 0; c < 2; c++) {
    const unsigned short* qp = &Q[(rowQ0 + lm) * DMODEL + hcol + c * 32 + lh * 8];
    qf[c] = ld_pair128(qp, qp + 16);
  }

  float mrun[8], lrun[8];
  v8f zero = {};
  v8f oacc[4];
#pragma unroll
  for (int r = 0; r < 8; r++) { mrun[r] = -1e30f; lrun[r] = 0.0f; }
#pragma unroll
  for (int t = 0; t < 4; t++) oacc[t] = zero;

  const float SC = 0.125f;  // 1/sqrt(HEADD)

  for (int kc = 0; kc < SEQL; kc += 32) {
    const size_t krow0 = (size_t)b * SEQL + kc;
    __syncthreads();  // previous chunk's Kt/Vt reads complete
    // stage K tile 32x64 row-major (coalesced dwords)
    for (int i = tid; i < 1024; i += 256) {
      int r = i >> 5, cp = i & 31;
      unsigned int val = *reinterpret_cast<const unsigned int*>(
          &Km[(krow0 + r) * DMODEL + hcol + cp * 2]);
      *reinterpret_cast<unsigned int*>(&Kt[r][cp * 2]) = val;
    }
    // stage V tile 32x64 transposed via 2x2 register transpose
    for (int i = tid; i < 512; i += 256) {
      int br = (i >> 5) * 2, bc = (i & 31) * 2;
      unsigned int d0 = *reinterpret_cast<const unsigned int*>(
          &Vm[(krow0 + br) * DMODEL + hcol + bc]);
      unsigned int d1 = *reinterpret_cast<const unsigned int*>(
          &Vm[(krow0 + br + 1) * DMODEL + hcol + bc]);
      unsigned int c0 = (d0 & 0xFFFFu) | (d1 << 16);
      unsigned int c1 = (d0 >> 16) | (d1 & 0xFFFF0000u);
      *reinterpret_cast<unsigned int*>(&Vt[bc][br])     = c0;
      *reinterpret_cast<unsigned int*>(&Vt[bc + 1][br]) = c1;
    }
    __syncthreads();

    // S(16x32) = Q(16x64) @ K^T(64x32): 2 N-tiles x 2 K-steps
    v8f s0 = zero, s1 = zero;
#pragma unroll
    for (int t = 0; t < 2; t++) {
#pragma unroll
      for (int c = 0; c < 2; c++) {
        const unsigned short* kp = &Kt[t * 16 + lm][c * 32 + lh * 8];
        v16us kfrag = ld_pair128(kp, kp + 16);
        if (t == 0) s0 = wmma_bf16(qf[c], kfrag, s0);
        else        s1 = wmma_bf16(qf[c], kfrag, s1);
      }
    }
    // online softmax: rows split across 16-lane halves; xor masks 1/2/4/8
    float alpha[8], p0[8], p1[8];
#pragma unroll
    for (int r = 0; r < 8; r++) {
      float v0 = s0[r] * SC, v1 = s1[r] * SC;
      float mx = fmaxf(v0, v1);
#pragma unroll
      for (int off2 = 1; off2 < 16; off2 <<= 1)
        mx = fmaxf(mx, __shfl_xor(mx, off2, 32));
      float mnew = fmaxf(mrun[r], mx);
      alpha[r] = expf(mrun[r] - mnew);
      mrun[r] = mnew;
      float e0 = expf(v0 - mnew), e1 = expf(v1 - mnew);
      p0[r] = e0; p1[r] = e1;
      float sm = e0 + e1;
#pragma unroll
      for (int off2 = 1; off2 < 16; off2 <<= 1)
        sm += __shfl_xor(sm, off2, 32);
      lrun[r] = lrun[r] * alpha[r] + sm;
    }
#pragma unroll
    for (int t = 0; t < 4; t++) {
#pragma unroll
      for (int r = 0; r < 8; r++) oacc[t][r] *= alpha[r];
    }
    // transpose P (C layout -> A layout) through per-wave LDS buffer
#pragma unroll
    for (int r = 0; r < 8; r++) {
      Plds[wid][lh * 8 + r][lm]      = f2bf(p0[r]);
      Plds[wid][lh * 8 + r][16 + lm] = f2bf(p1[r]);
    }
    __syncthreads();
    v16us pf = ld_pair128(&Plds[wid][lm][lh * 8], &Plds[wid][lm][16 + lh * 8]);
    // O(16x64) += P(16x32) @ V(32x64): 4 N-tiles
#pragma unroll
    for (int t = 0; t < 4; t++) {
      const unsigned short* vp = &Vt[t * 16 + lm][lh * 8];
      oacc[t] = wmma_bf16(pf, ld_pair128(vp, vp + 16), oacc[t]);
    }
  }
  // finalize: divide by row sums, store bf16
#pragma unroll
  for (int t = 0; t < 4; t++) {
    int c = hcol + t * 16 + lm;
#pragma unroll
    for (int r = 0; r < 8; r++) {
      float vv = oacc[t][r] / lrun[r];
      O[(rowQ0 + lh * 8 + r) * DMODEL + c] = f2bf(vv);
    }
  }
}

// ---------------------------------------------------------------------------
// Host launch
// ---------------------------------------------------------------------------
extern "C" void kernel_launch(void* const* d_in, const int* in_sizes, int n_in,
                              void* d_out, int out_size, void* d_ws, size_t ws_size,
                              hipStream_t stream) {
  (void)in_sizes; (void)n_in; (void)out_size; (void)ws_size;

  const float* x      = (const float*)d_in[0];
  const float* u      = (const float*)d_in[1];
  const float* t_emb  = (const float*)d_in[2];
  const float* ln1_g  = (const float*)d_in[3];
  const float* ln1_b  = (const float*)d_in[4];
  const float* aff1_w = (const float*)d_in[5];
  const float* aff1_b = (const float*)d_in[6];
  const float* ln2_g  = (const float*)d_in[7];
  const float* ln2_b  = (const float*)d_in[8];
  const float* aff2_w = (const float*)d_in[9];
  const float* aff2_b = (const float*)d_in[10];
  const float* sa_qw  = (const float*)d_in[11];
  const float* sa_qb  = (const float*)d_in[12];
  const float* sa_kw  = (const float*)d_in[13];
  const float* sa_kb  = (const float*)d_in[14];
  const float* sa_vw  = (const float*)d_in[15];
  const float* sa_vb  = (const float*)d_in[16];
  const float* sa_ow  = (const float*)d_in[17];
  const float* sa_ob  = (const float*)d_in[18];
  const float* ca_qw  = (const float*)d_in[19];
  const float* ca_qb  = (const float*)d_in[20];
  const float* ca_kw  = (const float*)d_in[21];
  const float* ca_kb  = (const float*)d_in[22];
  const float* ca_vw  = (const float*)d_in[23];
  const float* ca_vb  = (const float*)d_in[24];
  const float* ca_ow  = (const float*)d_in[25];
  const float* ca_ob  = (const float*)d_in[26];
  const float* ffn_w1 = (const float*)d_in[27];
  const float* ffn_b1 = (const float*)d_in[28];
  const float* ffn_w2 = (const float*)d_in[29];
  const float* ffn_b2 = (const float*)d_in[30];

  char* ws = (char*)d_ws;
  size_t off = 0;
  auto alloc = [&](size_t bytes) -> void* {
    void* p = ws + off;
    off += (bytes + 255) & ~(size_t)255;
    return p;
  };

  const size_t DD  = (size_t)DMODEL * DMODEL;
  const size_t DF  = (size_t)DMODEL * DFFN;
  const size_t MD  = (size_t)MTOT * DMODEL;

  unsigned short* w_sa_q = (unsigned short*)alloc(DD * 2);
  unsigned short* w_sa_k = (unsigned short*)alloc(DD * 2);
  unsigned short* w_sa_v = (unsigned short*)alloc(DD * 2);
  unsigned short* w_sa_o = (unsigned short*)alloc(DD * 2);
  unsigned short* w_ca_q = (unsigned short*)alloc(DD * 2);
  unsigned short* w_ca_k = (unsigned short*)alloc(DD * 2);
  unsigned short* w_ca_v = (unsigned short*)alloc(DD * 2);
  unsigned short* w_ca_o = (unsigned short*)alloc(DD * 2);
  unsigned short* w_ff1  = (unsigned short*)alloc(DF * 2);
  unsigned short* w_ff2  = (unsigned short*)alloc(DF * 2);
  unsigned short* u_bf   = (unsigned short*)alloc(MD * 2);
  unsigned short* xn_bf  = (unsigned short*)alloc(MD * 2);
  unsigned short* x_bf   = (unsigned short*)alloc(MD * 2);
  // q/k/v/ao are contiguous (each 8MB, 256-aligned) so they can be reused as
  // the 32MB FFN hidden buffer after cross-attention is done.
  unsigned short* q_bf   = (unsigned short*)alloc(MD * 2);
  unsigned short* k_bf   = (unsigned short*)alloc(MD * 2);
  unsigned short* v_bf   = (unsigned short*)alloc(MD * 2);
  unsigned short* ao_bf  = (unsigned short*)alloc(MD * 2);
  unsigned short* h_bf   = q_bf;  // [MTOT, DFFN] aliases q/k/v/ao
  float* x_f32 = (float*)alloc(MD * 4);
  float* aff1o = (float*)alloc((size_t)BATCH * 2 * DMODEL * 4);
  float* aff2o = (float*)alloc((size_t)BATCH * 2 * DMODEL * 4);

  // --- convert weights + u to bf16; copy x to working f32 buffer ---
  cvt_bf16_kernel<<<512, 256, 0, stream>>>(sa_qw, w_sa_q, (int)DD);
  cvt_bf16_kernel<<<512, 256, 0, stream>>>(sa_kw, w_sa_k, (int)DD);
  cvt_bf16_kernel<<<512, 256, 0, stream>>>(sa_vw, w_sa_v, (int)DD);
  cvt_bf16_kernel<<<512, 256, 0, stream>>>(sa_ow, w_sa_o, (int)DD);
  cvt_bf16_kernel<<<512, 256, 0, stream>>>(ca_qw, w_ca_q, (int)DD);
  cvt_bf16_kernel<<<512, 256, 0, stream>>>(ca_kw, w_ca_k, (int)DD);
  cvt_bf16_kernel<<<512, 256, 0, stream>>>(ca_vw, w_ca_v, (int)DD);
  cvt_bf16_kernel<<<512, 256, 0, stream>>>(ca_ow, w_ca_o, (int)DD);
  cvt_bf16_kernel<<<1024, 256, 0, stream>>>(ffn_w1, w_ff1, (int)DF);
  cvt_bf16_kernel<<<1024, 256, 0, stream>>>(ffn_w2, w_ff2, (int)DF);
  cvt_bf16_kernel<<<1024, 256, 0, stream>>>(u, u_bf, (int)MD);
  hipMemcpyAsync(x_f32, x, MD * sizeof(float), hipMemcpyDeviceToDevice, stream);

  // --- AdaLN modulation parameters ---
  affine_kernel<<<(BATCH * 2 * DMODEL) / 256, 256, 0, stream>>>(t_emb, aff1_w, aff1_b, aff1o);
  affine_kernel<<<(BATCH * 2 * DMODEL) / 256, 256, 0, stream>>>(t_emb, aff2_w, aff2_b, aff2o);

  dim3 gD(DMODEL / 128, MTOT / 128);
  dim3 gF(DFFN / 128, MTOT / 128);
  dim3 gA(SEQL / 128, BATCH * NHEADS);

  // --- AdaLN1 + self-attention (residual) ---
  adaln_kernel<<<MTOT, 256, 0, stream>>>(x_f32, ln1_g, ln1_b, aff1o, xn_bf);
  gemm_bf16_kernel<<<gD, 256, 0, stream>>>(xn_bf, w_sa_q, sa_qb, nullptr, nullptr, q_bf, MTOT, DMODEL, DMODEL, 0);
  gemm_bf16_kernel<<<gD, 256, 0, stream>>>(xn_bf, w_sa_k, sa_kb, nullptr, nullptr, k_bf, MTOT, DMODEL, DMODEL, 0);
  gemm_bf16_kernel<<<gD, 256, 0, stream>>>(xn_bf, w_sa_v, sa_vb, nullptr, nullptr, v_bf, MTOT, DMODEL, DMODEL, 0);
  flash_attn_kernel<<<gA, 256, 0, stream>>>(q_bf, k_bf, v_bf, ao_bf);
  gemm_bf16_kernel<<<gD, 256, 0, stream>>>(ao_bf, w_sa_o, sa_ob, x_f32, x_f32, x_bf, MTOT, DMODEL, DMODEL, 0);

  // --- cross-attention (q from x, k/v from u; residual) ---
  gemm_bf16_kernel<<<gD, 256, 0, stream>>>(x_bf, w_ca_q, ca_qb, nullptr, nullptr, q_bf, MTOT, DMODEL, DMODEL, 0);
  gemm_bf16_kernel<<<gD, 256, 0, stream>>>(u_bf, w_ca_k, ca_kb, nullptr, nullptr, k_bf, MTOT, DMODEL, DMODEL, 0);
  gemm_bf16_kernel<<<gD, 256, 0, stream>>>(u_bf, w_ca_v, ca_vb, nullptr, nullptr, v_bf, MTOT, DMODEL, DMODEL, 0);
  flash_attn_kernel<<<gA, 256, 0, stream>>>(q_bf, k_bf, v_bf, ao_bf);
  gemm_bf16_kernel<<<gD, 256, 0, stream>>>(ao_bf, w_ca_o, ca_ob, x_f32, x_f32, nullptr, MTOT, DMODEL, DMODEL, 0);

  // --- AdaLN2 + FFN (exact GELU; residual; final result -> d_out) ---
  adaln_kernel<<<MTOT, 256, 0, stream>>>(x_f32, ln2_g, ln2_b, aff2o, xn_bf);
  gemm_bf16_kernel<<<gF, 256, 0, stream>>>(xn_bf, w_ff1, ffn_b1, nullptr, nullptr, h_bf, MTOT, DFFN, DMODEL, 1);
  gemm_bf16_kernel<<<gD, 256, 0, stream>>>(h_bf, w_ff2, ffn_b2, x_f32, (float*)d_out, nullptr, MTOT, DMODEL, DFFN, 0);
}
